// SemanticDualHeadGAT_61538291417283
// MI455X (gfx1250) — matrix operand order repaired and checked by reading, not compiled, not run on hardware
//
#include <hip/hip_runtime.h>
#include <hip/hip_bf16.h>

#define N_NODES   25000
#define E_EDGES   400000
#define EA_EDGES  (E_EDGES + N_NODES)   // +self loops = 425000
#define HID       64
#define HEADS     4
#define HC        256                    // HEADS*HID
#define NODE_IN   128
#define EDGE_IN   32
#define NINT      12
#define FINK      160                    // 2*HID + EDGE_IN
#define NEG_SLOPE 0.2f

typedef __attribute__((ext_vector_type(16))) _Float16 v16h;
typedef __attribute__((ext_vector_type(4)))  _Float16 v4h;
typedef __attribute__((ext_vector_type(8)))  float    v8f;

// ---------------- WMMA helpers (wave32, 16x16x32 f16 -> f32) ----------------

__device__ __forceinline__ v8f wmma_f16(v16h a, v16h b, v8f c) {
    return __builtin_amdgcn_wmma_f32_16x16x32_f16(false, a, false, b, (short)0, c,
                                                  false, false);
}

// A fragment: 16x32 f16 tile, row-major in LDS with stride ld.
// Lanes 0-15 rows M=0..15; per-lane halves are one contiguous 16-half run
// (K = grp*16 + 0..15), so this lowers to 2x ds_load_b128.
__device__ __forceinline__ v16h frag_a(const _Float16* sA, int ld, int lane) {
    const int row  = lane & 15;
    const int grp8 = (lane >> 4) << 4;         // 0 or 16
    const _Float16* p = sA + row * ld;
    v16h a;
#pragma unroll
    for (int h = 0; h < 16; ++h) {
        int k = grp8 + h;   // grp*16 + h
        a[h] = p[k];
    }
    return a;
}

// ---- B fragments: pre-swizzled LDS layout ----
// Element (k,n) of a KxN matrix lives at:
//   ktile=k>>5, kk=k&31, ntile=n>>4, col=n&15, lane=((kk>>4)<<4)|col, h=kk&15
//   sB[(((ktile*numNt)+ntile)*32 + lane)*16 + h]
// so each lane's 16 halves are contiguous -> fragment load = 2x ds_load_b128.
__device__ __forceinline__ void stage_b_swz(const float* __restrict__ W,
                                            _Float16* __restrict__ sB,
                                            int K, int N, int tid, int nth) {
    const int numNt = N >> 4;
    for (int i = tid; i < K * N; i += nth) {
        int k = i / N, n = i - k * N;
        int ktile = k >> 5, kk = k & 31;
        int ntile = n >> 4, col = n & 15;
        int lane = ((kk >> 4) << 4) | col;
        int h    = kk & 15;
        sB[(((ktile * numNt) + ntile) * 32 + lane) * 16 + h] = (_Float16)W[i];
    }
}
__device__ __forceinline__ v16h frag_b_swz(const _Float16* sB, int numNt,
                                           int ktile, int ntile, int lane) {
    return *(const v16h*)&sB[(((ktile * numNt) + ntile) * 32 + lane) * 16];
}

// Ordered-uint encoding for float atomic max (monotone map f32 -> u32).
__device__ __forceinline__ unsigned enc_ord(float f) {
    unsigned u = __float_as_uint(f);
    return (u & 0x80000000u) ? ~u : (u | 0x80000000u);
}
__device__ __forceinline__ float dec_ord(unsigned u) {
    return __uint_as_float((u & 0x80000000u) ? (u ^ 0x80000000u) : ~u);
}
#define ENC_NEG_INF 0x007FFFFFu   // enc_ord(-inf)

// ---------------- utility kernels ----------------

__global__ void k_fill_f32(float* p, int n, float v) {
    for (int i = blockIdx.x * blockDim.x + threadIdx.x; i < n;
         i += gridDim.x * blockDim.x) p[i] = v;
}
__global__ void k_fill_u32(unsigned* p, int n, unsigned v) {
    for (int i = blockIdx.x * blockDim.x + threadIdx.x; i < n;
         i += gridDim.x * blockDim.x) p[i] = v;
}

__global__ void k_easum(const float* __restrict__ ea, float* __restrict__ easum) {
    int tid  = blockIdx.x * blockDim.x + threadIdx.x;
    int c    = tid & 31;
    int row0 = tid >> 5;
    int step = (gridDim.x * blockDim.x) >> 5;
    float s = 0.f;
    for (int r = row0; r < E_EDGES; r += step) s += ea[r * EDGE_IN + c];
    atomicAdd(&easum[c], s);
}
__global__ void k_eamean(const float* __restrict__ easum, float* __restrict__ eamean) {
    int c = threadIdx.x;
    if (c < EDGE_IN) eamean[c] = easum[c] * (1.f / (float)E_EDGES);
}

// ---------------- h = x @ emb_W + emb_b  (M=N, K=128, N=64) ----------------
// 128 threads = 4 waves; block computes 16 rows x 64 cols; wave -> one 16x16 tile.
__global__ void k_embed(const float* __restrict__ x, const float* __restrict__ W,
                        const float* __restrict__ b, float* __restrict__ hf,
                        _Float16* __restrict__ h16) {
    __shared__ alignas(32) _Float16 sA[16 * NODE_IN];     // 4 KB
    __shared__ alignas(32) _Float16 sB[NODE_IN * HID];    // 16 KB (swizzled)
    const int tid = threadIdx.x, wave = tid >> 5, lane = tid & 31;
    const int m0 = blockIdx.x * 16;
    stage_b_swz(W, sB, NODE_IN, HID, tid, blockDim.x);
    for (int i = tid * 4; i < 16 * NODE_IN; i += blockDim.x * 4) {
        int r = i >> 7, c = i & 127, gr = m0 + r;
        v4h hv = {};
        if (gr < N_NODES) {
            const float4 f = *(const float4*)&x[gr * NODE_IN + c];
            hv = (v4h){(_Float16)f.x, (_Float16)f.y, (_Float16)f.z, (_Float16)f.w};
        }
        *(v4h*)&sA[i] = hv;
    }
    __syncthreads();
    // issue all fragment loads, then chain WMMAs (pipelined dscnt waits)
    v16h af[4], bf[4];
#pragma unroll
    for (int kt = 0; kt < 4; ++kt) {
        af[kt] = frag_a(&sA[kt * 32], NODE_IN, lane);
        bf[kt] = frag_b_swz(sB, HID / 16, kt, wave, lane);
    }
    v8f acc = {};
#pragma unroll
    for (int kt = 0; kt < 4; ++kt) acc = wmma_f16(af[kt], bf[kt], acc);

    const int n0 = wave * 16;
    const int col = lane & 15, rb = (lane >> 4) << 3;
    const float bias = b[n0 + col];
    if (m0 + 16 <= N_NODES) {          // whole-tile fast path (no per-row branch)
#pragma unroll
        for (int r = 0; r < 8; ++r) {
            int gr = m0 + rb + r;
            float v = acc[r] + bias;
            hf[gr * HID + n0 + col]  = v;
            h16[gr * HID + n0 + col] = (_Float16)v;
        }
    } else {
#pragma unroll
        for (int r = 0; r < 8; ++r) {
            int gr = m0 + rb + r;
            if (gr < N_NODES) {
                float v = acc[r] + bias;
                hf[gr * HID + n0 + col]  = v;
                h16[gr * HID + n0 + col] = (_Float16)v;
            }
        }
    }
}

// ---------------- out = h @ W + b  (M=N, K=64, N=256) ----------------
// 256 threads = 8 waves; each wave does two 16-col tiles (8*16*2 = 256 cols).
__global__ void k_lin256(const _Float16* __restrict__ h16, const float* __restrict__ W,
                         const float* __restrict__ b, float* __restrict__ out) {
    __shared__ alignas(32) _Float16 sA[16 * HID];    // 2 KB
    __shared__ alignas(32) _Float16 sB[HID * HC];    // 32 KB (swizzled)
    const int tid = threadIdx.x, wave = tid >> 5, lane = tid & 31;
    const int m0 = blockIdx.x * 16;
    stage_b_swz(W, sB, HID, HC, tid, blockDim.x);
    for (int i = tid * 8; i < 16 * HID; i += blockDim.x * 8) {
        int r = i >> 6, c = i & 63, gr = m0 + r;
        uint4 z = {0, 0, 0, 0};
        *(uint4*)&sA[i] = (gr < N_NODES) ? *(const uint4*)&h16[gr * HID + c] : z;
    }
    __syncthreads();
    const int col = lane & 15, rb = (lane >> 4) << 3;
    const bool full = (m0 + 16 <= N_NODES);
    // A fragments shared by both column halves: load once
    v16h af[2];
#pragma unroll
    for (int kt = 0; kt < 2; ++kt) af[kt] = frag_a(&sA[kt * 32], HID, lane);
#pragma unroll
    for (int hp = 0; hp < 2; ++hp) {
        const int n0 = hp * 128 + wave * 16;
        v16h bf[2];
#pragma unroll
        for (int kt = 0; kt < 2; ++kt)
            bf[kt] = frag_b_swz(sB, HC / 16, kt, n0 >> 4, lane);
        v8f acc = {};
#pragma unroll
        for (int kt = 0; kt < 2; ++kt) acc = wmma_f16(af[kt], bf[kt], acc);
        const float bias = b[n0 + col];
        if (full) {
#pragma unroll
            for (int r = 0; r < 8; ++r)
                out[(m0 + rb + r) * HC + n0 + col] = acc[r] + bias;
        } else {
#pragma unroll
            for (int r = 0; r < 8; ++r) {
                int gr = m0 + rb + r;
                if (gr < N_NODES) out[gr * HC + n0 + col] = acc[r] + bias;
            }
        }
    }
}

// ---------------- fused edge pass 1 ----------------
// Per wave: 16 edges. ep = ea_tile(16x32,f16) @ We(32x256) via 16 WMMAs (K=32
// is exactly one step). m = leaky(xl[src]+xr[dst]+ep); logits = m . att via
// lane butterfly; store logits + ordered-uint atomic max per (dst, head).
__global__ void k_edge_logits(const float* __restrict__ xl, const float* __restrict__ xr,
                              const float* __restrict__ ea, const float* __restrict__ eamean,
                              const int* __restrict__ eidx,
                              const float* __restrict__ We, const float* __restrict__ att,
                              float* __restrict__ logits, unsigned* __restrict__ mxU) {
    __shared__ alignas(32) _Float16 sWe[EDGE_IN * HC];     // 16 KB (swizzled)
    __shared__ float    sAtt[HC];                          // att flat == channel idx
    __shared__ alignas(32) _Float16 sEA[8 * 16 * EDGE_IN]; // 8 KB, per-wave tiles
    __shared__ int      sS[8 * 16], sD[8 * 16];
    const int tid = threadIdx.x, wave = tid >> 5, lane = tid & 31;
    stage_b_swz(We, sWe, EDGE_IN, HC, tid, blockDim.x);
    for (int i = tid; i < HC; i += blockDim.x) sAtt[i] = att[i];
    __syncthreads();

    const int ntiles = (EA_EDGES + 15) / 16;
    for (int base = blockIdx.x * 8; base < ntiles; base += gridDim.x * 8) {
        const int t  = base + wave;
        const int e0 = t * 16;
        if (t < ntiles) {
            if (lane < 16) {
                int e = e0 + lane, s, d;
                if (e < E_EDGES)      { s = eidx[e]; d = eidx[E_EDGES + e]; }
                else if (e < EA_EDGES){ s = d = e - E_EDGES; }
                else                  { s = d = 0; }
                sS[wave * 16 + lane] = s; sD[wave * 16 + lane] = d;
            }
            for (int j = lane * 4; j < 16 * EDGE_IN; j += 32 * 4) {
                int r = j >> 5, c = j & 31, e = e0 + r;
                const float4 f = (e < E_EDGES) ? *(const float4*)&ea[e * EDGE_IN + c]
                                               : *(const float4*)&eamean[c];
                *(v4h*)&sEA[wave * 16 * EDGE_IN + j] =
                    (v4h){(_Float16)f.x, (_Float16)f.y, (_Float16)f.z, (_Float16)f.w};
            }
        }
        __syncthreads();

        const _Float16* myA = &sEA[wave * 16 * EDGE_IN];
        v16h afr = frag_a(myA, EDGE_IN, lane);   // one K=32 step == full EDGE_IN

        const int col = lane & 15, rb = (lane >> 4) << 3;
        int sI[8], dI[8];
#pragma unroll
        for (int r = 0; r < 8; ++r) {
            int e  = e0 + rb + r;
            int ok = (t < ntiles) && (e < EA_EDGES);
            sI[r] = ok ? sS[wave * 16 + rb + r] : 0;
            dI[r] = ok ? sD[wave * 16 + rb + r] : 0;
        }
        // prefetch the gathered rows (emits global_prefetch_b8)
#pragma unroll
        for (int r = 0; r < 8; ++r) {
            __builtin_prefetch(&xl[sI[r] * HC], 0, 3);
            __builtin_prefetch(&xr[dI[r] * HC], 0, 3);
        }

        float part[8][HEADS];
#pragma unroll
        for (int r = 0; r < 8; ++r)
#pragma unroll
            for (int h = 0; h < HEADS; ++h) part[r][h] = 0.f;

#pragma unroll
        for (int nt = 0; nt < 16; ++nt) {
            v16h bfr = frag_b_swz(sWe, HC / 16, 0, nt, lane);
            v8f  epz = {};
            v8f  ep  = wmma_f16(afr, bfr, epz);
            const int ch   = nt * 16 + col;
            const int head = nt >> 2;            // 4 tiles per head (64 ch)
            const float av = sAtt[ch];
#pragma unroll
            for (int r = 0; r < 8; ++r) {
                float m = xl[sI[r] * HC + ch] + xr[dI[r] * HC + ch] + ep[r];
                m = (m > 0.f) ? m : m * NEG_SLOPE;
                part[r][head] += m * av;
            }
        }
        // sum over the 16 lanes of each half-wave group (cols 0..15)
#pragma unroll
        for (int r = 0; r < 8; ++r)
#pragma unroll
            for (int h = 0; h < HEADS; ++h) {
                float v = part[r][h];
                v += __shfl_xor(v, 1, 16);
                v += __shfl_xor(v, 2, 16);
                v += __shfl_xor(v, 4, 16);
                v += __shfl_xor(v, 8, 16);
                part[r][h] = v;
            }
        if (t < ntiles && col < HEADS) {
#pragma unroll
            for (int r = 0; r < 8; ++r) {
                int e = e0 + rb + r;
                if (e < EA_EDGES) {
                    float lg = part[r][col];
                    logits[e * HEADS + col] = lg;
                    atomicMax(&mxU[dI[r] * HEADS + col], enc_ord(lg));
                }
            }
        }
        __syncthreads();
    }
}

// ---------------- edge pass 2: ex = exp(logit - max[dst]); den += ex ----------------
__global__ void k_edge_exp(float* __restrict__ lg, const unsigned* __restrict__ mxU,
                           float* __restrict__ den, const int* __restrict__ eidx) {
    const int total = EA_EDGES * HEADS;
    for (int i = blockIdx.x * blockDim.x + threadIdx.x; i < total;
         i += gridDim.x * blockDim.x) {
        int e = i >> 2, h = i & 3;
        int d = (e < E_EDGES) ? eidx[E_EDGES + e] : (e - E_EDGES);
        float mx = dec_ord(mxU[d * HEADS + h]);
        float ex = __expf(lg[i] - mx);
        lg[i] = ex;                              // overwrite logits with ex
        atomicAdd(&den[d * HEADS + h], ex);
    }
}

// ---------------- edge pass 3: out[dst] += xl[src] * alpha (4 ch / thread) ----------------
__global__ void k_edge_aggr(const float* __restrict__ ex, const float* __restrict__ den,
                            const float* __restrict__ xl, const int* __restrict__ eidx,
                            float* __restrict__ outacc) {
    const int total = EA_EDGES * (HC / 4);
    for (int i = blockIdx.x * blockDim.x + threadIdx.x; i < total;
         i += gridDim.x * blockDim.x) {
        int e = i >> 6, c4 = (i & 63) * 4, h = c4 >> 6;
        int s, d;
        if (e < E_EDGES) { s = eidx[e]; d = eidx[E_EDGES + e]; }
        else             { s = d = e - E_EDGES; }
        float alpha = ex[e * HEADS + h] / den[d * HEADS + h];
        const float4 v = *(const float4*)&xl[s * HC + c4];
        float* o = &outacc[d * HC + c4];
        atomicAdd(o + 0, v.x * alpha);
        atomicAdd(o + 1, v.y * alpha);
        atomicAdd(o + 2, v.z * alpha);
        atomicAdd(o + 3, v.w * alpha);
    }
}

// ---------------- node finalize: h = relu(mean_heads(out) + bias) ----------------
__global__ void k_node_final(const float* __restrict__ outacc, const float* __restrict__ bias,
                             float* __restrict__ hf, _Float16* __restrict__ h16) {
    int i = blockIdx.x * blockDim.x + threadIdx.x;      // over N * (HID/4)
    if (i >= N_NODES * (HID / 4)) return;
    int n = i >> 4, c4 = (i & 15) * 4;
    float acc[4] = {0.f, 0.f, 0.f, 0.f};
#pragma unroll
    for (int h = 0; h < HEADS; ++h) {
        const float4 t = *(const float4*)&outacc[n * HC + h * HID + c4];
        acc[0] += t.x; acc[1] += t.y; acc[2] += t.z; acc[3] += t.w;
    }
    const float4 bz = *(const float4*)&bias[c4];
    float4 o;
    o.x = fmaxf(acc[0] * 0.25f + bz.x, 0.f);
    o.y = fmaxf(acc[1] * 0.25f + bz.y, 0.f);
    o.z = fmaxf(acc[2] * 0.25f + bz.z, 0.f);
    o.w = fmaxf(acc[3] * 0.25f + bz.w, 0.f);
    *(float4*)&hf[n * HID + c4] = o;
    *(v4h*)&h16[n * HID + c4] =
        (v4h){(_Float16)o.x, (_Float16)o.y, (_Float16)o.z, (_Float16)o.w};
}

// ---------------- final edge head: relu[h[src]|h[dst]|ea] @ [int_W|imp_W] ----------------
// K=160 = five 16x16x32 WMMA steps; cols 0..11 = intent logits, col 12 = sigmoid.
// E is divisible by 16 so every tile is full.
__global__ void k_final(const float* __restrict__ hf, const float* __restrict__ ea,
                        const int* __restrict__ eidx,
                        const float* __restrict__ intW, const float* __restrict__ intB,
                        const float* __restrict__ impW, const float* __restrict__ impB,
                        float* __restrict__ outIntent, float* __restrict__ outImp) {
    __shared__ alignas(32) _Float16 sB[FINK * 16];      // 5 KB (swizzled, numNt=1)
    __shared__ alignas(32) _Float16 sA[8 * 16 * FINK];  // 40 KB, per-wave 16x160
    const int tid = threadIdx.x, wave = tid >> 5, lane = tid & 31;
    for (int i = tid; i < FINK * 16; i += blockDim.x) {
        int k = i >> 4, n = i & 15;
        float v = (n < NINT) ? intW[k * NINT + n] : ((n == NINT) ? impW[k] : 0.f);
        int kt = k >> 5, kk = k & 31;
        int ln = ((kk >> 4) << 4) | n;
        sB[(kt * 32 + ln) * 16 + (kk & 15)] = (_Float16)v;
    }
    __syncthreads();
    const int ntiles = E_EDGES / 16;
    for (int base = blockIdx.x * 8; base < ntiles; base += gridDim.x * 8) {
        const int t  = base + wave;
        const int e0 = t * 16;
        _Float16* myA = &sA[wave * 16 * FINK];
        if (t < ntiles) {
            for (int j = lane * 4; j < 16 * FINK; j += 32 * 4) {
                int r = j / FINK, c = j - r * FINK, e = e0 + r;
                float4 f;
                if (c < HID)          f = *(const float4*)&hf[eidx[e] * HID + c];
                else if (c < 2 * HID) f = *(const float4*)&hf[eidx[E_EDGES + e] * HID + (c - HID)];
                else                  f = *(const float4*)&ea[e * EDGE_IN + (c - 2 * HID)];
                *(v4h*)&myA[j] = (v4h){(_Float16)fmaxf(f.x, 0.f),
                                       (_Float16)fmaxf(f.y, 0.f),
                                       (_Float16)fmaxf(f.z, 0.f),
                                       (_Float16)fmaxf(f.w, 0.f)};
            }
        }
        __syncthreads();
        // issue all 20 ds_load_b128 up front, then chain 5 WMMAs
        v16h af[5], bf[5];
#pragma unroll
        for (int kt = 0; kt < 5; ++kt) {
            af[kt] = frag_a(&myA[kt * 32], FINK, lane);
            bf[kt] = frag_b_swz(sB, 1, kt, 0, lane);
        }
        v8f acc = {};
#pragma unroll
        for (int kt = 0; kt < 5; ++kt) acc = wmma_f16(af[kt], bf[kt], acc);

        const int col = lane & 15, rb = (lane >> 4) << 3;
        if (t < ntiles) {
            if (col < NINT) {
#pragma unroll
                for (int r = 0; r < 8; ++r)
                    outIntent[(e0 + rb + r) * NINT + col] = acc[r] + intB[col];
            } else if (col == NINT) {
                const float ib = impB[0];
#pragma unroll
                for (int r = 0; r < 8; ++r) {
                    float z = acc[r] + ib;
                    outImp[e0 + rb + r] = 1.f / (1.f + __expf(-z));
                }
            }
        }
        __syncthreads();
    }
}

// ---------------- host launcher ----------------

extern "C" void kernel_launch(void* const* d_in, const int* in_sizes, int n_in,
                              void* d_out, int out_size, void* d_ws, size_t ws_size,
                              hipStream_t stream) {
    (void)in_sizes; (void)n_in; (void)out_size; (void)ws_size;

    const float* x     = (const float*)d_in[0];
    const int*   eidx  = (const int*)d_in[1];       // (2,E): src=eidx[e], dst=eidx[E+e]
    const float* eatt  = (const float*)d_in[2];
    const float* emb_W = (const float*)d_in[3];
    const float* emb_b = (const float*)d_in[4];
    const float* g_Wl[2]   = {(const float*)d_in[5],  (const float*)d_in[12]};
    const float* g_bl[2]   = {(const float*)d_in[6],  (const float*)d_in[13]};
    const float* g_Wr[2]   = {(const float*)d_in[7],  (const float*)d_in[14]};
    const float* g_br[2]   = {(const float*)d_in[8],  (const float*)d_in[15]};
    const float* g_We[2]   = {(const float*)d_in[9],  (const float*)d_in[16]};
    const float* g_att[2]  = {(const float*)d_in[10], (const float*)d_in[17]};
    const float* g_bias[2] = {(const float*)d_in[11], (const float*)d_in[18]};
    const float* int_W = (const float*)d_in[19];
    const float* int_b = (const float*)d_in[20];
    const float* imp_W = (const float*)d_in[21];
    const float* imp_b = (const float*)d_in[22];

    char* wsb = (char*)d_ws;
    size_t off = 0;
    auto alloc = [&](size_t bytes) -> void* {
        void* p = wsb + off;
        off = (off + bytes + 255) & ~(size_t)255;
        return p;
    };
    float*     hf     = (float*)alloc((size_t)N_NODES * HID * 4);
    _Float16*  h16    = (_Float16*)alloc((size_t)N_NODES * HID * 2);
    float*     xl     = (float*)alloc((size_t)N_NODES * HC * 4);
    float*     xr     = (float*)alloc((size_t)N_NODES * HC * 4);
    float*     logb   = (float*)alloc((size_t)EA_EDGES * HEADS * 4);
    unsigned*  mxU    = (unsigned*)alloc((size_t)N_NODES * HEADS * 4);
    float*     den    = (float*)alloc((size_t)N_NODES * HEADS * 4);
    float*     outacc = (float*)alloc((size_t)N_NODES * HC * 4);
    float*     easum  = (float*)alloc(EDGE_IN * 4);
    float*     eamean = (float*)alloc(EDGE_IN * 4);

    float* outIntent = (float*)d_out;
    float* outImp    = outIntent + (size_t)E_EDGES * NINT;

    const int mblocks = (N_NODES + 15) / 16;                 // row-tile blocks
    const int eblocks = ((EA_EDGES + 15) / 16 + 7) / 8;      // 8 wave-tiles / block
    const int fblocks = (E_EDGES / 16 + 7) / 8;

    // edge_attr mean (for self-loop fill_value='mean')
    k_fill_f32<<<1, 64, 0, stream>>>(easum, EDGE_IN, 0.f);
    k_easum<<<256, 256, 0, stream>>>(eatt, easum);
    k_eamean<<<1, 32, 0, stream>>>(easum, eamean);

    // node embedding
    k_embed<<<mblocks, 128, 0, stream>>>(x, emb_W, emb_b, hf, h16);

    for (int L = 0; L < 2; ++L) {
        k_fill_u32<<<64, 256, 0, stream>>>(mxU, N_NODES * HEADS, ENC_NEG_INF);
        k_fill_f32<<<64, 256, 0, stream>>>(den, N_NODES * HEADS, 0.f);
        k_fill_f32<<<4096, 256, 0, stream>>>(outacc, N_NODES * HC, 0.f);

        k_lin256<<<mblocks, 256, 0, stream>>>(h16, g_Wl[L], g_bl[L], xl);
        k_lin256<<<mblocks, 256, 0, stream>>>(h16, g_Wr[L], g_br[L], xr);

        k_edge_logits<<<eblocks, 256, 0, stream>>>(xl, xr, eatt, eamean, eidx,
                                                   g_We[L], g_att[L], logb, mxU);
        k_edge_exp<<<2048, 256, 0, stream>>>(logb, mxU, den, eidx);
        k_edge_aggr<<<8192, 256, 0, stream>>>(logb, den, xl, eidx, outacc);
        k_node_final<<<(N_NODES * (HID / 4) + 255) / 256, 256, 0, stream>>>(outacc,
                                                                            g_bias[L],
                                                                            hf, h16);
    }

    k_final<<<fblocks, 256, 0, stream>>>(hf, eatt, eidx, int_W, int_b, imp_W, imp_b,
                                         outIntent, outImp);
}
